// CSAE_58909771432165
// MI455X (gfx1250) — compile-verified, matrix operand order checked by reading, not run on hardware
//
#include <hip/hip_runtime.h>
#include <hip/hip_bf16.h>
#include <stdint.h>

// ---------------------------------------------------------------------------
// CDNA5 / gfx1250 WMMA types
// ---------------------------------------------------------------------------
typedef __attribute__((ext_vector_type(16))) __bf16 v16bf;
typedef __attribute__((ext_vector_type(8)))  float  v8f;

#define GROUPS 256
#define BATCH  256
#define HW     64
#define ENC_C  1280
#define GPB    16      // groups per conv3 block

__device__ __forceinline__ __bf16 f2bf(float f) {
  uint32_t u = __builtin_bit_cast(uint32_t, f);
  uint32_t r = (u + 0x7fffu + ((u >> 16) & 1u)) >> 16;  // RNE
  return __builtin_bit_cast(__bf16, (uint16_t)r);
}
__device__ __forceinline__ float bf2f(__bf16 h) {
  uint32_t u = ((uint32_t)__builtin_bit_cast(uint16_t, h)) << 16;
  return __builtin_bit_cast(float, u);
}
__device__ __forceinline__ float lrelu(float v) { return v >= 0.f ? v : 0.01f * v; }

// Build a bf16 WMMA fragment (A 16xK or B NxK, K-contiguous rows, leading dim
// `ld`, even). lane&15 = row, lane>>4 = K-half per CDNA5 16-bit layout
// (ISA 7.12.2): VGPR v holds K = (v<4?0:16) + 8*(lane>>4) + 2*(v&3), +1.
// Lowered by the compiler to two ds_load_b128 per fragment.
__device__ __forceinline__ v16bf frag_lds(const __bf16* base, int ld, int k0) {
  const int lane  = threadIdx.x & 31;
  const __bf16* row = base + (lane & 15) * ld + k0;
  const int khalf = (lane >> 4) * 8;
  v16bf f;
#pragma unroll
  for (int v = 0; v < 8; ++v) {
    int kb = ((v < 4) ? 0 : 16) + khalf + (v & 3) * 2;
    uint32_t u = *(const uint32_t*)(row + kb);
    f[2 * v]     = __builtin_bit_cast(__bf16, (uint16_t)(u & 0xffffu));
    f[2 * v + 1] = __builtin_bit_cast(__bf16, (uint16_t)(u >> 16));
  }
  return f;
}

// ---------------------------------------------------------------------------
// fp32 -> bf16 convert (entry)
// ---------------------------------------------------------------------------
__global__ void k_f2bf(const float* __restrict__ s, __bf16* __restrict__ d, int n) {
  int i = blockIdx.x * blockDim.x + threadIdx.x;
  if (i < n) d[i] = f2bf(s[i]);
}

// ---------------------------------------------------------------------------
// grouped FC: per group g (256), Y[b,e] = act(sum_d X[b,g*D+d] W[g,e,d] + b).
// GEMM M=256(batch) N=K=D. Block = (g, 16-batch tile), 4 waves.
// A (16xD) staged once in LDS; B staged per 64x32 tile (coalesced fp32 load +
// bf16 convert); one wave per 16-wide N strip; wmma f32<-bf16.
// ---------------------------------------------------------------------------
template <bool RELU, bool F32OUT>
__global__ void k_gfc(const __bf16* __restrict__ x, const float* __restrict__ w,
                      const float* __restrict__ bias, void* __restrict__ outp, int D) {
  const int g    = blockIdx.x;
  const int mt   = blockIdx.y;
  const int wave = threadIdx.x >> 5;
  const int lane = threadIdx.x & 31;
  const int CHW  = GROUPS * D;

  __shared__ __align__(16) __bf16 ldsA[16 * 256];  // D <= 256 for all gfc layers
  __shared__ __align__(16) __bf16 ldsB[64 * 32];

  // stage A: 16 batches x D, d fastest -> coalesced
  for (int r = 0; r < 16; ++r)
    for (int d = threadIdx.x; d < D; d += blockDim.x)
      ldsA[r * D + d] = x[(size_t)(mt * 16 + r) * CHW + g * D + d];

  const float* wg = w + (size_t)g * D * D;
  const int nblocks = D >> 6;                     // 1..4 strips of 64 outputs
  for (int nb = 0; nb < nblocks; ++nb) {
    v8f acc = {};
    for (int k0 = 0; k0 < D; k0 += 32) {
      __syncthreads();
      // B[n][k] = W[g, nb*64+n, k0+k]; k fastest -> coalesced
      for (int idx = threadIdx.x; idx < 64 * 32; idx += blockDim.x) {
        int n = idx >> 5, k = idx & 31;
        ldsB[n * 32 + k] = f2bf(wg[(size_t)(nb * 64 + n) * D + k0 + k]);
      }
      if (k0 + 32 < D)
        __builtin_prefetch(wg + (size_t)(nb * 64) * D + k0 + 32, 0, 1);
      __syncthreads();
      v16bf a = frag_lds(ldsA, D, k0);
      v16bf b = frag_lds(ldsB + wave * 16 * 32, 32, 0);
      acc = __builtin_amdgcn_wmma_f32_16x16x32_bf16(false, a, false, b,
                                                    (short)0, acc, false, false);
    }
    const int n  = nb * 64 + wave * 16 + (lane & 15);
    const float bv = bias[g * D + n];
#pragma unroll
    for (int rr = 0; rr < 8; ++rr) {
      int b = mt * 16 + (lane >> 4) * 8 + rr;
      float v = acc[rr] + bv;
      if (RELU) v = lrelu(v);
      size_t o = (size_t)b * CHW + g * D + n;
      if (F32OUT) ((float*)outp)[o] = v;
      else        ((__bf16*)outp)[o] = f2bf(v);
    }
  }
}

// ---------------------------------------------------------------------------
// 1x1 conv: per batch b, Y[m,n] = act(sum_c X[c,m] W[n,c] + bias[n]).
// Block = batch b x 64-output strip: 64(M=hw) x 64(N) tile, K=C in 32-steps.
// Weights L2-resident across the 256 batches (192MB L2 >> 4MB panel).
// ---------------------------------------------------------------------------
__global__ void k_conv1(const __bf16* __restrict__ x, const float* __restrict__ w,
                        const float* __restrict__ bias, __bf16* __restrict__ out, int C) {
  const int nblk = blockIdx.x;
  const int b    = blockIdx.y;
  const int wave = threadIdx.x >> 5;
  const int lane = threadIdx.x & 31;

  __shared__ __align__(16) __bf16 ldsA[64 * 32];  // [m][k]
  __shared__ __align__(16) __bf16 ldsB[64 * 32];  // [n][k]

  v8f acc[4] = {};
  for (int k0 = 0; k0 < C; k0 += 32) {
    __syncthreads();
    for (int idx = threadIdx.x; idx < 64 * 32; idx += blockDim.x) {
      int m = idx & 63, k = idx >> 6;             // m fastest -> coalesced
      ldsA[m * 32 + k] = x[((size_t)b * C + (k0 + k)) * 64 + m];
    }
    for (int idx = threadIdx.x; idx < 64 * 32; idx += blockDim.x) {
      int n = idx >> 5, k = idx & 31;             // k fastest -> coalesced
      ldsB[n * 32 + k] = f2bf(w[(size_t)(nblk * 64 + n) * C + k0 + k]);
    }
    if (k0 + 32 < C)  // gfx1250 global_prefetch on the next weight K-slice
      __builtin_prefetch(w + (size_t)(nblk * 64) * C + k0 + 32, 0, 1);
    __syncthreads();

    v16bf a = frag_lds(ldsA + wave * 16 * 32, 32, 0);
#pragma unroll
    for (int nt = 0; nt < 4; ++nt) {
      v16bf bb = frag_lds(ldsB + nt * 16 * 32, 32, 0);
      acc[nt] = __builtin_amdgcn_wmma_f32_16x16x32_bf16(false, a, false, bb,
                                                        (short)0, acc[nt], false, false);
    }
  }
#pragma unroll
  for (int nt = 0; nt < 4; ++nt) {
    const int n  = nblk * 64 + nt * 16 + (lane & 15);
    const float bv = bias[n];
#pragma unroll
    for (int rr = 0; rr < 8; ++rr) {
      int m = wave * 16 + (lane >> 4) * 8 + rr;
      out[((size_t)b * C + n) * 64 + m] = f2bf(lrelu(acc[nt][rr] + bv));
    }
  }
}

// ---------------------------------------------------------------------------
// grouped 3x3 SAME conv (+bias+lrelu). K=9..45 -> VALU path (~7 GFLOP total).
// Block = (batch, 16 groups). The input slab is CONTIGUOUS bf16 in our
// channels-major layout, so it is DMA'd into LDS with CDNA5
// GLOBAL_LOAD_ASYNC_TO_LDS_B128 (ASYNCcnt, inline asm per ISA 15.18.3);
// weights for the block are staged in LDS (each reused by 64 spatial lanes).
// ---------------------------------------------------------------------------
__global__ void k_conv3(const __bf16* __restrict__ x, const float* __restrict__ w,
                        const float* __restrict__ bias, __bf16* __restrict__ out,
                        int Cin, int Cout, int cinpg, int coutpg) {
  const int gblk = blockIdx.x;                  // block of GPB groups
  const int b    = blockIdx.y;
  const int cinBlk  = GPB * cinpg;              // <= 80 input channels
  const int coutBlk = GPB * coutpg;             // <= 80 output channels

  __shared__ __align__(16) __bf16 sin[GPB * 5 * 64];       // <= 10 KB
  __shared__ __align__(16) float  swt[GPB * 5 * 5 * 9];    // <= 14.4 KB

  // async DMA: global (contiguous) -> LDS, 16B per lane per issue
  {
    const __bf16* src = x + ((size_t)b * Cin + (size_t)gblk * cinBlk) * 64;
    const int bytes = cinBlk * 64 * 2;
    uint32_t lbase = (uint32_t)(uintptr_t)(&sin[0]);
    for (int ofs = threadIdx.x * 16; ofs < bytes; ofs += blockDim.x * 16) {
      uint64_t ga = (uint64_t)(uintptr_t)((const char*)src + ofs);
      uint32_t la = lbase + (uint32_t)ofs;
      asm volatile("global_load_async_to_lds_b128 %0, %1, off"
                   :: "v"(la), "v"(ga) : "memory");
    }
  }
  // weights: coalesced fp32 copy (overlaps with the async DMA above)
  {
    const int wn = coutBlk * cinpg * 9;
    const float* wsrc = w + (size_t)gblk * coutBlk * cinpg * 9;
    for (int i = threadIdx.x; i < wn; i += blockDim.x) swt[i] = wsrc[i];
  }
  asm volatile("s_wait_asynccnt 0x0" ::: "memory");
  __syncthreads();

  for (int oidx = threadIdx.x; oidx < coutBlk * 64; oidx += blockDim.x) {
    int m  = oidx & 63;
    int og = oidx >> 6;                         // local output channel
    int o  = gblk * coutBlk + og;
    int gl = og / coutpg;                       // local group
    int yy = m >> 3, xx = m & 7;
    float acc = bias[o];
    const float* wo = swt + og * cinpg * 9;
    for (int ic = 0; ic < cinpg; ++ic) {
      const __bf16* xc = sin + (gl * cinpg + ic) * 64;
      const float*  wi = wo + ic * 9;
#pragma unroll
      for (int ky = 0; ky < 3; ++ky) {
        int sy = yy + ky - 1;
        if (sy < 0 || sy > 7) continue;
#pragma unroll
        for (int kx = 0; kx < 3; ++kx) {
          int sx = xx + kx - 1;
          if (sx < 0 || sx > 7) continue;
          acc += bf2f(xc[sy * 8 + sx]) * wi[ky * 3 + kx];
        }
      }
    }
    out[((size_t)b * Cout + o) * 64 + m] = f2bf(lrelu(acc));
  }
}

// ---------------------------------------------------------------------------
// 2x2 maxpool + unpool: keep max, zero the rest. Writes fp32 `encoded` output
// and a bf16 copy for the decoder in one pass.
// ---------------------------------------------------------------------------
__global__ void k_pool_unpool(const __bf16* __restrict__ e, float* __restrict__ enc_f32,
                              __bf16* __restrict__ enc_bf) {
  int idx = blockIdx.x * blockDim.x + threadIdx.x;
  int total = BATCH * ENC_C * 16;
  if (idx >= total) return;
  int blk = idx & 15;
  int t   = idx >> 4;               // b*C + c
  int by = blk >> 2, bx = blk & 3;
  size_t base = (size_t)t * 64;
  int off[4]; float v[4];
#pragma unroll
  for (int j = 0; j < 4; ++j) {
    int dy = j >> 1, dx = j & 1;
    off[j] = (2 * by + dy) * 8 + (2 * bx + dx);
    v[j]   = bf2f(e[base + off[j]]);
  }
  int am = 0; float mv = v[0];
#pragma unroll
  for (int j = 1; j < 4; ++j) if (v[j] > mv) { mv = v[j]; am = j; }
#pragma unroll
  for (int j = 0; j < 4; ++j) {
    float o = (j == am) ? v[j] : 0.f;
    enc_f32[base + off[j]] = o;
    enc_bf [base + off[j]] = f2bf(o);
  }
}

// ---------------------------------------------------------------------------
// driver
// ---------------------------------------------------------------------------
extern "C" void kernel_launch(void* const* d_in, const int* in_sizes, int n_in,
                              void* d_out, int out_size, void* d_ws, size_t ws_size,
                              hipStream_t stream) {
  (void)in_sizes; (void)n_in; (void)out_size; (void)ws_size;
  static const int C[5] = {256, 512, 768, 1024, 1280};

  const float* x = (const float*)d_in[0];
  const size_t ACT_BYTES = (size_t)BATCH * 1280 * HW * sizeof(__bf16);  // 40 MiB
  __bf16* bufA = (__bf16*)d_ws;
  __bf16* bufB = (__bf16*)((char*)d_ws + ((ACT_BYTES + 255) & ~(size_t)255));
  __bf16* cur = bufA;
  __bf16* nxt = bufB;

  const dim3 blk128(128), blk256(256);
  auto swap = [&]() { __bf16* t = cur; cur = nxt; nxt = t; };

  { int n = BATCH * 256 * HW;
    k_f2bf<<<dim3((n + 255) / 256), blk256, 0, stream>>>(x, bufA, n); }

  // ---- encoder pre-stages ----
  for (int i = 0; i < 4; ++i) {
    int ci = C[i], co = C[i + 1], D = ci / 4;
    const float* gw = (const float*)d_in[1 + 6 * i];
    const float* gb = (const float*)d_in[2 + 6 * i];
    const float* w1 = (const float*)d_in[3 + 6 * i];
    const float* b1 = (const float*)d_in[4 + 6 * i];
    const float* w3 = (const float*)d_in[5 + 6 * i];
    const float* b3 = (const float*)d_in[6 + 6 * i];

    k_gfc<true, false><<<dim3(GROUPS, 16), blk128, 0, stream>>>(cur, gw, gb, (void*)nxt, D);
    swap();
    k_conv1<<<dim3(ci / 64, BATCH), blk128, 0, stream>>>(cur, w1, b1, nxt, ci);
    swap();
    k_conv3<<<dim3(GROUPS / GPB, BATCH), blk256, 0, stream>>>(cur, w3, b3, nxt,
                                                              ci, co, ci / GROUPS, co / GROUPS);
    swap();
  }

  // ---- bottleneck: enc conv3, pool/unpool, dec conv3 ----
  {
    const float* enc_w = (const float*)d_in[25];
    const float* enc_b = (const float*)d_in[26];
    const float* dec_w = (const float*)d_in[27];
    const float* dec_b = (const float*)d_in[28];
    k_conv3<<<dim3(GROUPS / GPB, BATCH), blk256, 0, stream>>>(cur, enc_w, enc_b, nxt,
                                                              ENC_C, ENC_C, 5, 5);
    swap();
    int np = BATCH * ENC_C * 16;
    k_pool_unpool<<<dim3((np + 255) / 256), blk256, 0, stream>>>(cur, (float*)d_out, nxt);
    swap();
    k_conv3<<<dim3(GROUPS / GPB, BATCH), blk256, 0, stream>>>(cur, dec_w, dec_b, nxt,
                                                              ENC_C, ENC_C, 5, 5);
    swap();
  }

  // ---- decoder post-stages ----
  for (int i = 0; i < 4; ++i) {
    int ri = C[4 - i], ro = C[3 - i], Do = ro / 4;
    const float* w3 = (const float*)d_in[29 + 6 * i];
    const float* b3 = (const float*)d_in[30 + 6 * i];
    const float* w1 = (const float*)d_in[31 + 6 * i];
    const float* b1 = (const float*)d_in[32 + 6 * i];
    const float* gw = (const float*)d_in[33 + 6 * i];
    const float* gb = (const float*)d_in[34 + 6 * i];

    k_conv3<<<dim3(GROUPS / GPB, BATCH), blk256, 0, stream>>>(cur, w3, b3, nxt,
                                                              ri, ro, ri / GROUPS, ro / GROUPS);
    swap();
    k_conv1<<<dim3(ro / 64, BATCH), blk128, 0, stream>>>(cur, w1, b1, nxt, ro);
    swap();
    if (i != 3) {
      k_gfc<true, false><<<dim3(GROUPS, 16), blk128, 0, stream>>>(cur, gw, gb, (void*)nxt, Do);
      swap();
    } else {
      float* hout = (float*)d_out + (size_t)BATCH * ENC_C * HW;
      k_gfc<false, true><<<dim3(GROUPS, 16), blk128, 0, stream>>>(cur, gw, gb, (void*)hout, Do);
    }
  }
}
